// EncodeProcessDecode_62749472195069
// MI455X (gfx1250) — compile-verified, matrix operand order checked by reading, not run on hardware
//
#include <hip/hip_runtime.h>

typedef __bf16 bf16;
typedef __attribute__((ext_vector_type(16))) __bf16 v16bf;
typedef __attribute__((ext_vector_type(8)))  float  v8f;

#define NN     100000
#define NE     1600000
#define NT_N   (NN / 16)   // 6250 node tiles
#define NT_E   (NE / 16)   // 100000 edge tiles
#define NPASS  5

// ---------------- fragment helpers ----------------

__device__ __forceinline__ v8f zero8() {
  v8f z = {0.f, 0.f, 0.f, 0.f, 0.f, 0.f, 0.f, 0.f};
  return z;
}

__device__ __forceinline__ v8f wmma_bf(v16bf a, v16bf b, v8f c) {
  // D = A(16x32 bf16) x B(32x16 bf16) + C(16x16 f32)
  return __builtin_amdgcn_wmma_f32_16x16x32_bf16(false, a, false, b, (short)0, c,
                                                 false, false);
}

// A-fragment (16x32, bf16) for lane `lane`, from a row-major bf16 row pointer.
// Lane layout (ISA 7.12.2): row = lane&15, K = kb..kb+7 and 16+kb..16+kb+7,
// kb = (lane>>4)*8  -> two contiguous 16B chunks.
__device__ __forceinline__ v16bf afrag_bf16(const bf16* rowp, int koff, int lane) {
  int kb = (lane >> 4) << 3;
  const bf16* p = rowp + koff + kb;
  union U { v16bf v; int4 q[2]; } u;
  u.q[0] = *(const int4*)(p);
  u.q[1] = *(const int4*)(p + 16);
  return u.v;
}

// Same, but converting from an f32 source row (used for the aggregated messages).
__device__ __forceinline__ v16bf afrag_f32(const float* rowp, int koff, int lane) {
  int kb = (lane >> 4) << 3;
  const float* p = rowp + koff + kb;
  v16bf a;
#pragma unroll
  for (int i = 0; i < 8; ++i) {
    a[i]     = (bf16)p[i];
    a[8 + i] = (bf16)p[16 + i];
  }
  return a;
}

// B-fragment from pre-swizzled weights: layout [K/32][4 ntiles][32 lanes][16 elems],
// so each lane reads its 32 bytes contiguously (2x b128).
__device__ __forceinline__ v16bf bfrag(const bf16* wsw, int ks, int nt, int lane) {
  const bf16* p = wsw + ((size_t)((((ks << 2) + nt) << 5) + lane) << 4);
  union U { v16bf v; int4 q[2]; } u;
  u.q[0] = ((const int4*)p)[0];
  u.q[1] = ((const int4*)p)[1];
  return u.v;
}

// ---------------- weight swizzle: f32 row-major [K,64] -> bf16 fragment order ----

__global__ __launch_bounds__(256) void k_swizzle(const float* __restrict__ W,
                                                 bf16* __restrict__ dst, int total) {
  int tid = blockIdx.x * 256 + threadIdx.x;
  if (tid >= total) return;
  int e    = tid & 15;
  int lane = (tid >> 4) & 31;
  int nt   = (tid >> 9) & 3;
  int ks   = tid >> 11;
  int kk = ((lane >> 4) << 3) + (e & 7) + ((e >> 3) << 4);
  int k  = (ks << 5) + kk;
  int n  = (nt << 4) + (lane & 15);
  dst[tid] = (bf16)W[k * 64 + n];
}

// ---------------- node encoder: [2]->relu[64]->[64] ----------------

__global__ __launch_bounds__(256) void k_enc_node(
    const float* __restrict__ X, const float* __restrict__ W1,
    const float* __restrict__ b1, const bf16* __restrict__ W2s,
    const float* __restrict__ b2, bf16* __restrict__ out) {
  __shared__ bf16 h1[128 * 64];
  int t = threadIdx.x;
  int nb0 = blockIdx.x * 128;
#pragma unroll
  for (int i = 0; i < 8192; i += 256) {
    int idx = t + i;
    int nl = idx >> 6, f = idx & 63;
    int n = nb0 + nl;
    float hv = 0.f;
    if (n < NN) {
      float x0 = X[n * 2 + 0], x1 = X[n * 2 + 1];
      hv = fmaxf(x0 * W1[f] + x1 * W1[64 + f] + b1[f], 0.f);
    }
    h1[idx] = (bf16)hv;
  }
  __syncthreads();
  int wave = t >> 5, lane = t & 31;
  int tile = blockIdx.x * 8 + wave;
  if (tile >= NT_N) return;
  const bf16* rowp = &h1[((wave << 4) + (lane & 15)) << 6];
  v16bf a0 = afrag_bf16(rowp, 0, lane);
  v16bf a1 = afrag_bf16(rowp, 32, lane);
  int nbase = tile << 4;
  int m0 = (lane >> 4) << 3, c = lane & 15;
#pragma unroll
  for (int nt = 0; nt < 4; ++nt) {
    v8f acc = zero8();
    acc = wmma_bf(a0, bfrag(W2s, 0, nt, lane), acc);
    acc = wmma_bf(a1, bfrag(W2s, 1, nt, lane), acc);
    float bb = b2[(nt << 4) + c];
#pragma unroll
    for (int vv = 0; vv < 8; ++vv)
      out[(size_t)(nbase + m0 + vv) * 64 + (nt << 4) + c] = (bf16)(acc[vv] + bb);
  }
}

// ---------------- edge encoder: [1]->relu[64]->[64] ----------------
// Each block handles 8 chunks of 128 edges; the 8 W2 B-fragments stay resident
// in VGPRs across the whole block lifetime.

__global__ __launch_bounds__(256) void k_enc_edge(
    const float* __restrict__ Ein, const float* __restrict__ W1,
    const float* __restrict__ b1, const bf16* __restrict__ W2s,
    const float* __restrict__ b2, bf16* __restrict__ out) {
  __shared__ bf16 h1[128 * 64];
  int t = threadIdx.x;
  int wave = t >> 5, lane = t & 31;
  int m0 = (lane >> 4) << 3, c = lane & 15;

  v16bf bw[2][4];
#pragma unroll
  for (int ks = 0; ks < 2; ++ks)
#pragma unroll
    for (int nt = 0; nt < 4; ++nt) bw[ks][nt] = bfrag(W2s, ks, nt, lane);

#pragma unroll 1
  for (int cchunk = 0; cchunk < 8; ++cchunk) {
    long chunk = (long)blockIdx.x * 8 + cchunk;
    long eb0 = chunk * 128;
    if (eb0 >= NE) break;  // uniform per block
    __syncthreads();       // protect h1 reuse across iterations
#pragma unroll
    for (int i = 0; i < 8192; i += 256) {
      int idx = t + i;
      int el = idx >> 6, f = idx & 63;
      long e = eb0 + el;
      float hv = 0.f;
      if (e < NE) hv = fmaxf(Ein[e] * W1[f] + b1[f], 0.f);
      h1[idx] = (bf16)hv;
    }
    __syncthreads();
    long tile = chunk * 8 + wave;
    if (tile < NT_E) {
      const bf16* rowp = &h1[((wave << 4) + c) << 6];
      v16bf a0 = afrag_bf16(rowp, 0, lane);
      v16bf a1 = afrag_bf16(rowp, 32, lane);
      long ebase = tile << 4;
#pragma unroll
      for (int nt = 0; nt < 4; ++nt) {
        v8f acc = zero8();
        acc = wmma_bf(a0, bw[0][nt], acc);
        acc = wmma_bf(a1, bw[1][nt], acc);
        float bb = b2[(nt << 4) + c];
#pragma unroll
        for (int vv = 0; vv < 8; ++vv)
          out[(ebase + m0 + vv) * 64 + (nt << 4) + c] = (bf16)(acc[vv] + bb);
      }
    }
  }
}

// ---------------- messages: gather senders, [E,128]@[128,64], scatter-add ------
// Each wave keeps all 16 Wmsg B-fragments resident in VGPRs and streams 8
// consecutive edge tiles through them, amortizing weight loads 8x.

__global__ __launch_bounds__(256) void k_msg(
    const bf16* __restrict__ nodes, const bf16* __restrict__ ef,
    const int* __restrict__ send, const int* __restrict__ recv,
    const bf16* __restrict__ Wms, float* __restrict__ agg) {
  int wave = threadIdx.x >> 5, lane = threadIdx.x & 31;
  long tile0 = ((long)blockIdx.x * 8 + wave) * 8;
  if (tile0 >= NT_E) return;
  int r15 = lane & 15, m0 = (lane >> 4) << 3;

  v16bf bw[4][4];
#pragma unroll
  for (int ks = 0; ks < 4; ++ks)
#pragma unroll
    for (int nt = 0; nt < 4; ++nt) bw[ks][nt] = bfrag(Wms, ks, nt, lane);

#pragma unroll 1
  for (int tloop = 0; tloop < 8; ++tloop) {
    long tile = tile0 + tloop;
    if (tile >= NT_E) break;
    long ebase = tile << 4;
    int srow = send[ebase + r15];
    const bf16* sp = nodes + (size_t)srow * 64;
    const bf16* ep = ef + (size_t)(ebase + r15) * 64;
    v16bf a0 = afrag_bf16(sp, 0, lane);   // sender feats K 0..31
    v16bf a1 = afrag_bf16(sp, 32, lane);  // sender feats K 32..63
    v16bf a2 = afrag_bf16(ep, 0, lane);   // edge feats   K 64..95
    v16bf a3 = afrag_bf16(ep, 32, lane);  // edge feats   K 96..127
    int ridx[8];
#pragma unroll
    for (int vv = 0; vv < 8; ++vv) ridx[vv] = recv[ebase + m0 + vv];
#pragma unroll
    for (int nt = 0; nt < 4; ++nt) {
      v8f acc = zero8();
      acc = wmma_bf(a0, bw[0][nt], acc);
      acc = wmma_bf(a1, bw[1][nt], acc);
      acc = wmma_bf(a2, bw[2][nt], acc);
      acc = wmma_bf(a3, bw[3][nt], acc);
#pragma unroll
      for (int vv = 0; vv < 8; ++vv)
        unsafeAtomicAdd(&agg[(size_t)ridx[vv] * 64 + (nt << 4) + r15], acc[vv]);
    }
  }
}

// ---------------- node update: MLP + residual Wnode + LayerNorm ----------------

__global__ __launch_bounds__(256) void k_node(
    bf16* __restrict__ nodes, const float* __restrict__ agg,
    const bf16* __restrict__ Wns, const bf16* __restrict__ W1s,
    const float* __restrict__ b1, const bf16* __restrict__ W2s,
    const float* __restrict__ b2, const float* __restrict__ lns,
    const float* __restrict__ lnb) {
  __shared__ bf16  hbuf[8][16 * 64];
  __shared__ float ybuf[8][16 * 65];  // pad to 65 to dodge bank conflicts
  int wave = threadIdx.x >> 5, lane = threadIdx.x & 31;
  int tile = blockIdx.x * 8 + wave;
  if (tile >= NT_N) return;
  int nbase = tile << 4;
  int r15 = lane & 15, m0 = (lane >> 4) << 3;
  bf16* nrow = nodes + (size_t)(nbase + r15) * 64;
  const float* arow = agg + (size_t)(nbase + r15) * 64;
  v16bf an0 = afrag_bf16(nrow, 0, lane);
  v16bf an1 = afrag_bf16(nrow, 32, lane);
  v16bf ag0 = afrag_f32(arow, 0, lane);
  v16bf ag1 = afrag_f32(arow, 32, lane);

  // h = relu(concat(nodes, agg) @ W1 + b1)
  v8f h[4];
#pragma unroll
  for (int nt = 0; nt < 4; ++nt) {
    v8f acc = zero8();
    acc = wmma_bf(an0, bfrag(W1s, 0, nt, lane), acc);
    acc = wmma_bf(an1, bfrag(W1s, 1, nt, lane), acc);
    acc = wmma_bf(ag0, bfrag(W1s, 2, nt, lane), acc);
    acc = wmma_bf(ag1, bfrag(W1s, 3, nt, lane), acc);
    float bb = b1[(nt << 4) + r15];
#pragma unroll
    for (int vv = 0; vv < 8; ++vv) acc[vv] = fmaxf(acc[vv] + bb, 0.f);
    h[nt] = acc;
  }
  // D-layout -> A-layout transpose of h through per-wave LDS
#pragma unroll
  for (int nt = 0; nt < 4; ++nt)
#pragma unroll
    for (int vv = 0; vv < 8; ++vv)
      hbuf[wave][((m0 + vv) << 6) + (nt << 4) + r15] = (bf16)h[nt][vv];
  asm volatile("s_wait_dscnt 0" ::: "memory");
  const bf16* hrow = &hbuf[wave][r15 << 6];
  v16bf ah0 = afrag_bf16(hrow, 0, lane);
  v16bf ah1 = afrag_bf16(hrow, 32, lane);

  // y = nodes@Wnode + h@W2 + b2
#pragma unroll
  for (int nt = 0; nt < 4; ++nt) {
    v8f acc = zero8();
    acc = wmma_bf(an0, bfrag(Wns, 0, nt, lane), acc);
    acc = wmma_bf(an1, bfrag(Wns, 1, nt, lane), acc);
    acc = wmma_bf(ah0, bfrag(W2s, 0, nt, lane), acc);
    acc = wmma_bf(ah1, bfrag(W2s, 1, nt, lane), acc);
    float bb = b2[(nt << 4) + r15];
#pragma unroll
    for (int vv = 0; vv < 8; ++vv)
      ybuf[wave][(m0 + vv) * 65 + (nt << 4) + r15] = acc[vv] + bb;
  }
  asm volatile("s_wait_dscnt 0" ::: "memory");

  // LayerNorm over H=64: 2 lanes per row, combine halves with shfl_xor(16)
  int hh = lane >> 4;
  float s = 0.f, ss = 0.f;
#pragma unroll
  for (int cc = 0; cc < 32; ++cc) {
    float x = ybuf[wave][r15 * 65 + (hh << 5) + cc];
    s += x;
    ss += x * x;
  }
  s  += __shfl_xor(s, 16);
  ss += __shfl_xor(ss, 16);
  float mu  = s * (1.f / 64.f);
  float var = ss * (1.f / 64.f) - mu * mu;
  float inv = rsqrtf(var + 1e-6f);
#pragma unroll
  for (int cc = 0; cc < 32; ++cc) {
    int col = (hh << 5) + cc;
    float x = (ybuf[wave][r15 * 65 + col] - mu) * inv * lns[col] + lnb[col];
    nodes[(size_t)(nbase + r15) * 64 + col] = (bf16)x;
  }
}

// ---------------- decoder: relu(nodes@W1+b1) @ W2 + b2 -> [N,1] ----------------

__global__ __launch_bounds__(256) void k_dec(
    const bf16* __restrict__ nodes, const bf16* __restrict__ W1s,
    const float* __restrict__ b1, const float* __restrict__ W2,
    const float* __restrict__ b2, float* __restrict__ out) {
  int wave = threadIdx.x >> 5, lane = threadIdx.x & 31;
  int tile = blockIdx.x * 8 + wave;
  if (tile >= NT_N) return;
  int nbase = tile << 4, r15 = lane & 15, m0 = (lane >> 4) << 3;
  const bf16* nrow = nodes + (size_t)(nbase + r15) * 64;
  v16bf a0 = afrag_bf16(nrow, 0, lane);
  v16bf a1 = afrag_bf16(nrow, 32, lane);
  float part[8];
#pragma unroll
  for (int vv = 0; vv < 8; ++vv) part[vv] = 0.f;
#pragma unroll
  for (int nt = 0; nt < 4; ++nt) {
    v8f acc = zero8();
    acc = wmma_bf(a0, bfrag(W1s, 0, nt, lane), acc);
    acc = wmma_bf(a1, bfrag(W1s, 1, nt, lane), acc);
    float bb = b1[(nt << 4) + r15];
    float w2 = W2[(nt << 4) + r15];
#pragma unroll
    for (int vv = 0; vv < 8; ++vv) part[vv] += fmaxf(acc[vv] + bb, 0.f) * w2;
  }
#pragma unroll
  for (int md = 1; md < 16; md <<= 1)
#pragma unroll
    for (int vv = 0; vv < 8; ++vv) part[vv] += __shfl_xor(part[vv], md);
  if (r15 == 0) {
    float bb = b2[0];
#pragma unroll
    for (int vv = 0; vv < 8; ++vv) out[nbase + m0 + vv] = part[vv] + bb;
  }
}

// ---------------- host orchestration ----------------

extern "C" void kernel_launch(void* const* d_in, const int* in_sizes, int n_in,
                              void* d_out, int out_size, void* d_ws, size_t ws_size,
                              hipStream_t stream) {
  (void)in_sizes; (void)n_in; (void)out_size; (void)ws_size;

  // setup_inputs() insertion order, params pytree in insertion order:
  const float* X      = (const float*)d_in[0];
  const float* Eraw   = (const float*)d_in[1];
  const int*   send   = (const int*)d_in[2];
  const int*   recv   = (const int*)d_in[3];
  const float* enN_W1 = (const float*)d_in[4];
  const float* enN_b1 = (const float*)d_in[5];
  const float* enN_W2 = (const float*)d_in[6];
  const float* enN_b2 = (const float*)d_in[7];
  const float* enE_W1 = (const float*)d_in[8];
  const float* enE_b1 = (const float*)d_in[9];
  const float* enE_W2 = (const float*)d_in[10];
  const float* enE_b2 = (const float*)d_in[11];
  const float* dec_W1 = (const float*)d_in[12];
  const float* dec_b1 = (const float*)d_in[13];
  const float* dec_W2 = (const float*)d_in[14];
  const float* dec_b2 = (const float*)d_in[15];

  // workspace layout
  char*  ws     = (char*)d_ws;
  bf16*  nodesB = (bf16*)(ws);                                   // NN*64 bf16
  bf16*  edgesB = (bf16*)(ws + (size_t)NN * 64 * 2);             // NE*64 bf16
  float* agg    = (float*)(ws + (size_t)NN * 64 * 2 + (size_t)NE * 64 * 2);
  bf16*  wsw    = (bf16*)(ws + (size_t)NN * 64 * 2 + (size_t)NE * 64 * 2 +
                          (size_t)NN * 64 * 4);

  bf16* enN_W2s = wsw + 0;
  bf16* enE_W2s = wsw + 4096;
  bf16* dec_W1s = wsw + 8192;

  auto sw = [&](const float* W, bf16* dst, int K) {
    int total = K * 64;
    k_swizzle<<<(total + 255) / 256, 256, 0, stream>>>(W, dst, total);
  };

  sw(enN_W2, enN_W2s, 64);
  sw(enE_W2, enE_W2s, 64);
  sw(dec_W1, dec_W1s, 64);

  bf16* pWmsg[NPASS]; bf16* pW1[NPASS]; bf16* pW2[NPASS]; bf16* pWn[NPASS];
  const float *pb1[NPASS], *pb2[NPASS], *plns[NPASS], *plnb[NPASS];
  for (int p = 0; p < NPASS; ++p) {
    int base = 16 + p * 8;
    const float* Wmsg = (const float*)d_in[base + 0];
    const float* nmW1 = (const float*)d_in[base + 1];
    pb1[p]            = (const float*)d_in[base + 2];
    const float* nmW2 = (const float*)d_in[base + 3];
    pb2[p]            = (const float*)d_in[base + 4];
    const float* Wn   = (const float*)d_in[base + 5];
    plns[p]           = (const float*)d_in[base + 6];
    plnb[p]           = (const float*)d_in[base + 7];
    bf16* pbase = wsw + 12288 + p * 24576;
    pWmsg[p] = pbase;          sw(Wmsg, pWmsg[p], 128);
    pW1[p]   = pbase + 8192;   sw(nmW1, pW1[p], 128);
    pW2[p]   = pbase + 16384;  sw(nmW2, pW2[p], 64);
    pWn[p]   = pbase + 20480;  sw(Wn,   pWn[p], 64);
  }

  const int nodeBlocks = (NT_N + 7) / 8;     // 782
  const int encEBlocks = (NE + 1023) / 1024; // 1563  (8 chunks of 128/block)
  const int msgBlocks  = (NT_E + 63) / 64;   // 1563  (8 waves x 8 tiles/block)

  k_enc_node<<<nodeBlocks, 256, 0, stream>>>(X, enN_W1, enN_b1, enN_W2s, enN_b2,
                                             nodesB);
  k_enc_edge<<<encEBlocks, 256, 0, stream>>>(Eraw, enE_W1, enE_b1, enE_W2s,
                                             enE_b2, edgesB);

  for (int p = 0; p < NPASS; ++p) {
    (void)hipMemsetAsync(agg, 0, (size_t)NN * 64 * 4, stream);
    k_msg<<<msgBlocks, 256, 0, stream>>>(nodesB, edgesB, send, recv, pWmsg[p],
                                         agg);
    k_node<<<nodeBlocks, 256, 0, stream>>>(nodesB, agg, pWn[p], pW1[p], pb1[p],
                                           pW2[p], pb2[p], plns[p], plnb[p]);
  }

  k_dec<<<nodeBlocks, 256, 0, stream>>>(nodesB, dec_W1s, dec_b1, dec_W2, dec_b2,
                                        (float*)d_out);
}
